// W8A16LinearLayer_49392123904762
// MI455X (gfx1250) — matrix operand order repaired and checked by reading, not compiled, not run on hardware
//
#include <hip/hip_runtime.h>

typedef float    v4f  __attribute__((ext_vector_type(4)));
typedef int      v4i  __attribute__((ext_vector_type(4)));
typedef _Float16 v4h  __attribute__((ext_vector_type(4)));
typedef _Float16 v8h  __attribute__((ext_vector_type(8)));
typedef _Float16 v16h __attribute__((ext_vector_type(16)));
typedef float    v8f  __attribute__((ext_vector_type(8)));

#define BM 128
#define BN 256
#define BK 32
#define LDT 40      // LDS row stride in halfs: 32 + 8 pad = 80B -> conflict-free b128 reads
#define THREADS 256

__global__ __launch_bounds__(THREADS) void w8a16_wmma(
    const float* __restrict__ A,   // [M, K] fp32 activations
    const int*   __restrict__ W,   // [N, K] int8-valued weights stored as int32
    const float* __restrict__ scl, // [N] per-output scales
    const float* __restrict__ bia, // [N] bias
    float* __restrict__ O,         // [M, N] fp32 out
    int M, int N, int K)
{
  __shared__ __align__(16) _Float16 sA[2][BM * LDT];   // 10 KB per buffer
  __shared__ __align__(16) _Float16 sB[2][BN * LDT];   // 20 KB per buffer

  const int t    = threadIdx.x;
  const int wid  = t >> 5;         // 8 waves (wave32)
  const int lane = t & 31;
  const int wm   = wid & 1;        // wave M index (0..1) -> 64 rows each
  const int wn   = wid >> 1;       // wave N index (0..3) -> 64 cols each
  const int lr   = lane & 15;
  const int hi   = lane >> 4;      // half-wave select
  const int kbA  = hi << 3;        // A fragment K base: 0 / 8
  const int kbB  = hi << 3;        // B fragment K base: 0 / 8

  const int m0 = blockIdx.y * BM;
  const int n0 = blockIdx.x * BN;
  const int steps = K / BK;

  // ---- prologue: K-tile 0 -> buffer 0 (global -> convert fp16 -> LDS) ----
  {
    // A tile: 128 rows x 8 chunks = 1024 chunks -> 4 per thread
#pragma unroll
    for (int i = 0; i < 4; ++i) {
      const int idx = i * THREADS + t;
      const int row = idx >> 3;
      const int c4  = (idx & 7) << 2;
      v4f a = *(const v4f*)(A + (size_t)(m0 + row) * K + c4);
      *(v4h*)(&sA[0][row * LDT + c4]) = __builtin_convertvector(a, v4h);
    }
    // B tile: 256 rows x 8 chunks = 2048 chunks -> 8 per thread
#pragma unroll
    for (int i = 0; i < 8; ++i) {
      const int idx = i * THREADS + t;
      const int row = idx >> 3;
      const int c4  = (idx & 7) << 2;
      v4i w = *(const v4i*)(W + (size_t)(n0 + row) * K + c4);
      *(v4h*)(&sB[0][row * LDT + c4]) = __builtin_convertvector(w, v4h);
    }
  }
  __syncthreads();

  v8f acc[4][4] = {};

  for (int kt = 0; kt < steps; ++kt) {
    const int buf = kt & 1;

    // ---- stage next K-tile into registers (overlaps with WMMA below) ----
    v4f ra[4];
    v4i rw[8];
    const bool more = (kt + 1) < steps;
    if (more) {
      const int k0 = (kt + 1) * BK;
#pragma unroll
      for (int i = 0; i < 4; ++i) {
        const int idx = i * THREADS + t;
        const int row = idx >> 3;
        const int c4  = (idx & 7) << 2;
        ra[i] = *(const v4f*)(A + (size_t)(m0 + row) * K + k0 + c4);
      }
#pragma unroll
      for (int i = 0; i < 8; ++i) {
        const int idx = i * THREADS + t;
        const int row = idx >> 3;
        const int c4  = (idx & 7) << 2;
        rw[i] = *(const v4i*)(W + (size_t)(n0 + row) * K + k0 + c4);
      }
    }

    // ---- compute on current buffer: 4x4 tiles, 16 v_wmma per K-step ----
    const _Float16* a_tile = sA[buf];
    const _Float16* b_tile = sB[buf];

    v16h bf[4];
#pragma unroll
    for (int nt = 0; nt < 4; ++nt) {
      // B 32x16 fragment: lane = column n, K = [kbB,kbB+8) and [kbB+16,kbB+24)
      const _Float16* p = b_tile + (wn * 64 + nt * 16 + lr) * LDT + kbB;
      union { v16h v; v8h h[2]; } u;
      u.h[0] = *(const v8h*)(p);
      u.h[1] = *(const v8h*)(p + 16);
      bf[nt] = u.v;
    }
#pragma unroll
    for (int mt = 0; mt < 4; ++mt) {
      // A 16x32 fragment: lane = row m, K = [kbA,kbA+8) and [kbA+16,kbA+24)
      const _Float16* p = a_tile + (wm * 64 + mt * 16 + lr) * LDT + kbA;
      union { v16h v; v8h h[2]; } u;
      u.h[0] = *(const v8h*)(p);
      u.h[1] = *(const v8h*)(p + 16);
      const v16h af = u.v;
#pragma unroll
      for (int nt = 0; nt < 4; ++nt) {
        acc[mt][nt] = __builtin_amdgcn_wmma_f32_16x16x32_f16(
            false, af, false, bf[nt], (short)0, acc[mt][nt], false, false);
      }
    }

    // ---- write staged regs into the other buffer, then barrier ----
    if (more) {
#pragma unroll
      for (int i = 0; i < 4; ++i) {
        const int idx = i * THREADS + t;
        const int row = idx >> 3;
        const int c4  = (idx & 7) << 2;
        *(v4h*)(&sA[buf ^ 1][row * LDT + c4]) = __builtin_convertvector(ra[i], v4h);
      }
#pragma unroll
      for (int i = 0; i < 8; ++i) {
        const int idx = i * THREADS + t;
        const int row = idx >> 3;
        const int c4  = (idx & 7) << 2;
        *(v4h*)(&sB[buf ^ 1][row * LDT + c4]) = __builtin_convertvector(rw[i], v4h);
      }
    }
    __syncthreads();
  }

  // ---- epilogue: per-output-column scale + bias ----
  // C/D layout: lane(0..15)=N col, VGPR r = M row (lanes 16..31: M row + 8)
#pragma unroll
  for (int nt = 0; nt < 4; ++nt) {
    const int n = n0 + wn * 64 + nt * 16 + lr;
    const float s = scl[n];
    const float b = bia[n];
#pragma unroll
    for (int mt = 0; mt < 4; ++mt) {
      const int mb = m0 + wm * 64 + mt * 16 + (hi << 3);
#pragma unroll
      for (int r = 0; r < 8; ++r) {
        O[(size_t)(mb + r) * N + n] = acc[mt][nt][r] * s + b;
      }
    }
  }
}

extern "C" void kernel_launch(void* const* d_in, const int* in_sizes, int n_in,
                              void* d_out, int out_size, void* d_ws, size_t ws_size,
                              hipStream_t stream) {
  const float* A   = (const float*)d_in[0];  // inputs [B,S,D_IN] fp32
  const int*   W   = (const int*)d_in[1];    // int8_weights [D_OUT,D_IN] int32
  const float* scl = (const float*)d_in[2];  // scales [D_OUT]
  const float* bia = (const float*)d_in[3];  // bias [1,D_OUT]
  float* out = (float*)d_out;

  const int N = in_sizes[2];           // D_OUT = 4096
  const int K = in_sizes[1] / N;       // D_IN  = 4096
  const int M = in_sizes[0] / K;       // B*S   = 16384

  dim3 grid(N / BN, M / BM);           // (16, 128)
  w8a16_wmma<<<grid, THREADS, 0, stream>>>(A, W, scl, bia, out, M, N, K);
}
